// VanillaSFNN_86474871538198
// MI455X (gfx1250) — compile-verified, matrix operand order checked by reading, not run on hardware
//
#include <hip/hip_runtime.h>

// ---------------------------------------------------------------------------
// Spiking FNN: out = scan_LIF(x @ W_in^T + b_in) @ W_out^T + b_out
//   x: (256,1000,700) f32   W_in: (200,700)   W_out: (35,200)
//
// Primary path (needs ~107MB workspace):
//   k0: W_in f32 -> bf16, padded to 208x704
//   k1: bf16-WMMA GEMM with W resident in LDS (296KB of the 320KB pool) and
//       a double-buffered A tile; 2-deep software pipeline on B fragments
//   k2: parallel-over-(b,h) LIF scan over T, spike counts -> workspace
//   k3: tiny output GEMM
// Fallback path (small workspace): fully fused kernel, W_in resident in LDS.
// ---------------------------------------------------------------------------

typedef __bf16 bf16;
typedef __attribute__((ext_vector_type(16))) __bf16 v16bf;
typedef __attribute__((ext_vector_type(8)))  __bf16 v8bf;
typedef __attribute__((ext_vector_type(8)))  float  v8f;
typedef __attribute__((ext_vector_type(2)))  float  f32x2;

constexpr int B_ = 256, T_ = 1000, D_ = 700, H_ = 200, O_ = 35;
constexpr int HP  = 208;        // H padded to 13 tiles of 16
constexpr int NT  = 13;         // N tiles
constexpr int KPD = 704;        // K padded to 22 steps of 32
constexpr int KSTEPS = 22;
constexpr long M_TOTAL = (long)B_ * T_;   // 256000 GEMM rows

// --- helpers ---------------------------------------------------------------
__device__ __forceinline__ unsigned pack_bf16x2(float a, float b) {
    union { bf16 h[2]; unsigned u; } pk;
    pk.h[0] = (bf16)a; pk.h[1] = (bf16)b;
    return pk.u;
}
__device__ __forceinline__ v16bf cat16(v8bf lo, v8bf hi) {
    return __builtin_shufflevector(lo, hi, 0,1,2,3,4,5,6,7,8,9,10,11,12,13,14,15);
}
__device__ __forceinline__ v16bf load_frag(const bf16* p) {
    return cat16(*(const v8bf*)p, *(const v8bf*)(p + 16));
}

// ---------------------------------------------------------------------------
// k0: convert W_in (200x700 f32) -> wbf (208x704 bf16, zero padded)
// ---------------------------------------------------------------------------
__global__ __launch_bounds__(256)
void snn_wconv(const float* __restrict__ Win, bf16* __restrict__ wbf) {
    int u = blockIdx.x * 256 + threadIdx.x;           // pair index
    if (u >= HP * (KPD / 2)) return;
    int h = u / (KPD / 2), j = u - h * (KPD / 2);
    f32x2 w = {0.f, 0.f};
    if (h < H_ && j < D_ / 2) w = *(const f32x2*)(Win + h * D_ + 2 * j);
    *(unsigned*)&wbf[h * KPD + 2 * j] = pack_bf16x2(w.x, w.y);
}

// ---------------------------------------------------------------------------
// k1: GEMM  i_all[m, n] = sum_k x[m,k] * W[n,k]   (m over B*T, n over HP)
// block = 256 thr (8 waves); block tile = 128 m-rows x all 208 n.
// W (208 x stride 712 bf16 = 296KB) staged into LDS ONCE per block.
// A double-buffered in LDS per 32-wide k-step (2 x 128 x stride 40 = 20KB).
// Next A chunk's 8 global f32x2 loads issue before the 13 WMMAs; ds_store
// drains loadcnt after the math. B frags: 2-deep rotating pipeline so 4 LDS
// loads stay in flight (steady-state s_wait_dscnt <= 2, not 0).
// ---------------------------------------------------------------------------
constexpr int MB   = 128;       // m rows per block
constexpr int WST  = 712;       // W LDS row stride (356 dw: 16 lanes->16 banks)
constexpr int AST  = 40;        // A LDS row stride (20 dw: 64 banks covered)
constexpr int NA   = MB * 16 / 256;       // 8 A f32-pairs per thread per step
constexpr size_t GEMM_LDS = (size_t)HP * WST * sizeof(bf16)
                          + 2u * MB * AST * sizeof(bf16);   // 316,672 B

__global__ __launch_bounds__(256)
void snn_gemm1(const float* __restrict__ x, const bf16* __restrict__ wbf,
               bf16* __restrict__ iall) {
    extern __shared__ char smem[];
    bf16* Wl = (bf16*)smem;                 // HP x WST
    bf16* Ab = Wl + (size_t)HP * WST;       // 2 x (MB x AST)

    const int tid  = threadIdx.x;
    const int wv   = tid >> 5;
    const int lane = tid & 31;
    const int col  = lane & 15;
    const int hf   = lane >> 4;
    const long m0  = (long)blockIdx.x * MB;

    v8f acc[NT] = {};

    // --- stage W once: (stride 704 global) -> (stride 712 LDS) ---
    for (int u = tid; u < HP * (KPD / 2); u += 256) {
        int h = u / (KPD / 2), j = u - h * (KPD / 2);
        *(unsigned*)&Wl[h * WST + 2 * j] = *(const unsigned*)&wbf[h * KPD + 2 * j];
    }

    // --- A staging helpers (one 32-wide k-step per chunk) ---
    auto loadA = [&](int c, f32x2 (&areg)[NA]) {
        #pragma unroll
        for (int i = 0; i < NA; ++i) {
            int u = tid + i * 256;
            int r = u >> 4, j = u & 15;
            int kp = c * 16 + j;                       // global f32-pair index
            areg[i] = (f32x2){0.f, 0.f};
            if (kp < D_ / 2) {
                const float* src = x + (m0 + r) * D_ + 2 * kp;
                areg[i] = *(const f32x2*)src;
                if (kp + 16 < D_ / 2) __builtin_prefetch(src + 32, 0, 0);
            }
        }
    };
    auto storeA = [&](int buf, const f32x2 (&areg)[NA]) {
        #pragma unroll
        for (int i = 0; i < NA; ++i) {
            int u = tid + i * 256;
            int r = u >> 4, j = u & 15;
            *(unsigned*)&Ab[buf * (MB * AST) + r * AST + 2 * j] =
                pack_bf16x2(areg[i].x, areg[i].y);
        }
    };

    // --- one k-step: 13 WMMAs sharing one A frag; B frags 2-deep pipeline ---
    auto compute = [&](int c, int buf) {
        const bf16* arow  = Ab + buf * (MB * AST) + (wv * 16 + col) * AST + hf * 8;
        const bf16* bbase = Wl + col * WST + c * 32 + hf * 8;
        v16bf av = load_frag(arow);
        v16bf bfr[3];
        bfr[0] = load_frag(bbase);
        bfr[1] = load_frag(bbase + 16 * WST);
        #pragma unroll
        for (int n = 0; n < NT; ++n) {
            if (n + 2 < NT)
                bfr[(n + 2) % 3] = load_frag(bbase + (n + 2) * 16 * WST);
            acc[n] = __builtin_amdgcn_wmma_f32_16x16x32_bf16(
                false, av, false, bfr[n % 3], (short)0, acc[n], false, false);
        }
    };

    // --- pipelined main loop over 22 k-steps ---
    {
        f32x2 areg[NA];
        loadA(0, areg);
        storeA(0, areg);
    }
    __syncthreads();
    #pragma unroll 1
    for (int c = 0; c < KSTEPS; ++c) {
        f32x2 areg[NA];
        const bool pf = (c + 1 < KSTEPS);
        if (pf) loadA(c + 1, areg);                    // global loads in flight
        compute(c, c & 1);                             // 13 WMMAs hide latency
        if (pf) storeA((c + 1) & 1, areg);             // drains loadcnt here
        __syncthreads();
    }

    // C layout: lane<16 -> M=vgpr, N=lane; lane>=16 -> M=vgpr+8, N=lane-16
    const long mrow = m0 + wv * 16 + hf * 8;
    #pragma unroll
    for (int n = 0; n < NT; ++n)
        #pragma unroll
        for (int r = 0; r < 8; ++r)
            iall[(mrow + r) * HP + n * 16 + col] = (bf16)acc[n][r];
}

// ---------------------------------------------------------------------------
// k2: LIF scan over T. One block per batch b; thread = neuron h (7 waves).
// ---------------------------------------------------------------------------
__global__ __launch_bounds__(224)
void snn_scan(const bf16* __restrict__ iall, const float* __restrict__ bin,
              const float* __restrict__ taup, float* __restrict__ counts) {
    const int b = blockIdx.x, h = threadIdx.x;
    if (h >= H_) return;
    const float alpha = 1.f / (1.f + __expf(-taup[0]));
    const float beta  = 1.f - alpha;
    const float bh    = bin[h];
    const bf16* p = iall + (size_t)b * T_ * HP + h;
    float v = 0.f, c = 0.f;
    #pragma unroll 4
    for (int t = 0; t < T_; ++t) {
        float it = (float)p[(size_t)t * HP] + bh;
        v = alpha * v + beta * it;
        float s = (v >= 1.0f) ? 1.f : 0.f;
        c += s;
        v -= s;
    }
    counts[b * H_ + h] = c;
}

// ---------------------------------------------------------------------------
// k3: out[b,o] = counts[b,:] . W_out[o,:] + b_out[o]
// ---------------------------------------------------------------------------
__global__ __launch_bounds__(256)
void snn_outg(const float* __restrict__ counts, const float* __restrict__ Wout,
              const float* __restrict__ bout, float* __restrict__ out) {
    int u = blockIdx.x * 256 + threadIdx.x;
    if (u >= B_ * O_) return;
    int b = u / O_, o = u - b * O_;
    float a = bout[o];
    const float* cr = counts + b * H_;
    const float* wr = Wout + o * H_;
    #pragma unroll 4
    for (int h = 0; h < H_; ++h) a += cr[h] * wr[h];
    out[u] = a;
}

// ---------------------------------------------------------------------------
// Fallback: fully fused. 16 blocks x 13 waves; W_in bf16 resident in LDS
// (208 x 712 stride = 290KB of the 320KB WGP LDS); per timestep stage 16
// batch rows of x into LDS, each wave WMMAs its 16-neuron tile (2-deep
// pipelined A/B fragment loads), LIF in regs.
// ---------------------------------------------------------------------------
constexpr int KPF = 712;                   // fused LDS k-stride (bank friendly)
constexpr int FTHREADS = 416;              // 13 waves

__global__ __launch_bounds__(FTHREADS)
void snn_fused(const float* __restrict__ x, const float* __restrict__ Win,
               const float* __restrict__ bin, const float* __restrict__ Wout,
               const float* __restrict__ bout, const float* __restrict__ taup,
               float* __restrict__ out) {
    extern __shared__ char smem[];
    bf16*  wlds = (bf16*)smem;             // HP x KPF
    bf16*  xlds = wlds + HP * KPF;         // 16 x KPF
    float* clds = (float*)xlds;            // aliases x tile: 16 x HP f32

    const int tid  = threadIdx.x;
    const int wv   = tid >> 5;
    const int lane = tid & 31;
    const int col  = lane & 15;
    const int hf   = lane >> 4;
    const int b0   = blockIdx.x * 16;
    const int hb   = wv * 16;

    {   // zero pads once
        unsigned* p = (unsigned*)smem;
        for (int i = tid; i < (HP * KPF + 16 * KPF) / 2; i += FTHREADS) p[i] = 0u;
    }
    __syncthreads();
    for (int u = tid; u < H_ * (D_ / 2); u += FTHREADS) {
        int h = u / (D_ / 2), j = u - h * (D_ / 2);
        f32x2 w = *(const f32x2*)(Win + h * D_ + 2 * j);
        *(unsigned*)&wlds[h * KPF + 2 * j] = pack_bf16x2(w.x, w.y);
    }
    __syncthreads();

    const float alpha = 1.f / (1.f + __expf(-taup[0]));
    const float beta  = 1.f - alpha;
    const int   hcol  = hb + col;
    const float bh    = (hcol < H_) ? bin[hcol] : 0.f;

    v8f v = {}, cnt = {};
    for (int t = 0; t < T_; ++t) {
        for (int u = tid; u < 16 * (D_ / 2); u += FTHREADS) {
            int r = u / (D_ / 2), j = u - r * (D_ / 2);
            const float* src = x + ((size_t)(b0 + r) * T_ + t) * D_ + 2 * j;
            f32x2 xv = *(const f32x2*)src;
            if (t + 1 < T_) __builtin_prefetch(src + D_, 0, 0);  // next timestep
            *(unsigned*)&xlds[r * KPF + 2 * j] = pack_bf16x2(xv.x, xv.y);
        }
        __syncthreads();

        v8f acc = {};
        const bf16* arow = xlds + col * KPF + hf * 8;
        const bf16* brow = wlds + (hb + col) * KPF + hf * 8;
        v16bf av[3], bv[3];
        av[0] = load_frag(arow);       bv[0] = load_frag(brow);
        av[1] = load_frag(arow + 32);  bv[1] = load_frag(brow + 32);
        #pragma unroll
        for (int kt = 0; kt < KSTEPS; ++kt) {
            if (kt + 2 < KSTEPS) {
                av[(kt + 2) % 3] = load_frag(arow + (kt + 2) * 32);
                bv[(kt + 2) % 3] = load_frag(brow + (kt + 2) * 32);
            }
            acc = __builtin_amdgcn_wmma_f32_16x16x32_bf16(
                false, av[kt % 3], false, bv[kt % 3], (short)0, acc, false, false);
        }
        #pragma unroll
        for (int r = 0; r < 8; ++r) {
            float it = acc[r] + bh;
            float vv = alpha * v[r] + beta * it;
            float s  = (vv >= 1.0f) ? 1.f : 0.f;
            cnt[r] += s;
            v[r] = vv - s;
        }
        __syncthreads();
    }

    #pragma unroll
    for (int r = 0; r < 8; ++r)
        clds[(hf * 8 + r) * HP + hb + col] = cnt[r];
    __syncthreads();

    for (int u = tid; u < 16 * O_; u += FTHREADS) {
        int r = u / O_, o = u - r * O_;
        float a = bout[o];
        const float* cr = clds + r * HP;
        const float* wr = Wout + o * H_;
        for (int h = 0; h < H_; ++h) a += cr[h] * wr[h];
        out[(b0 + r) * O_ + o] = a;
    }
}

// ---------------------------------------------------------------------------
extern "C" void kernel_launch(void* const* d_in, const int* in_sizes, int n_in,
                              void* d_out, int out_size, void* d_ws, size_t ws_size,
                              hipStream_t stream) {
    const float* x    = (const float*)d_in[0];
    const float* Win  = (const float*)d_in[1];
    const float* bin  = (const float*)d_in[2];
    const float* Wout = (const float*)d_in[3];
    const float* bout = (const float*)d_in[4];
    const float* tau  = (const float*)d_in[5];
    float* out = (float*)d_out;

    const size_t wbytes = (size_t)HP * KPD * sizeof(bf16);        //   0.29 MB
    const size_t ibytes = (size_t)M_TOTAL * HP * sizeof(bf16);    // 106.5  MB
    const size_t cbytes = (size_t)B_ * H_ * sizeof(float);        //   0.20 MB

    if (ws_size >= wbytes + ibytes + cbytes) {
        bf16*  wbf    = (bf16*)d_ws;
        bf16*  iall   = (bf16*)((char*)d_ws + wbytes);
        float* counts = (float*)((char*)d_ws + wbytes + ibytes);

        snn_wconv<<<(HP * (KPD / 2) + 255) / 256, 256, 0, stream>>>(Win, wbf);
        snn_gemm1<<<(int)(M_TOTAL / MB), 256, GEMM_LDS, stream>>>(x, wbf, iall);
        snn_scan<<<B_, 224, 0, stream>>>(iall, bin, tau, counts);
        snn_outg<<<(B_ * O_ + 255) / 256, 256, 0, stream>>>(counts, Wout, bout, out);
    } else {
        const size_t shmem = (size_t)(HP * KPF + 16 * KPF) * sizeof(bf16); // 311KB
        snn_fused<<<B_ / 16, FTHREADS, shmem, stream>>>(x, Win, bin, Wout,
                                                        bout, tau, out);
    }
}
